// NonLocal_Direct_1589137899829
// MI455X (gfx1250) — matrix operand order repaired and checked by reading, not compiled, not run on hardware
//
#include <hip/hip_runtime.h>
#include <stdint.h>

// NonLocal self-attention block, MI455X (gfx1250, wave32, WMMA).
// B=8, C=256, N=48*48=2304.
//   g = x viewed as [N,C] row-major per batch (raw reshape of [C,H,W]!)
//   S = g g^T ; P = softmax(S) ; O = P g        (flash-attention streaming)
//   mask[o,n] = sum_c W[o,c] O[n,c] ; final = mask + x
// bf16 WMMA (16x16x32) with f32 accumulation; scores never hit memory.

#define BATCH 8
#define CH    256
#define NSP   2304
#define CNE   (CH * NSP)          // 589824 elems per batch
#define QTILE 48                  // query rows per workgroup (16 per wave)
#define KTILE 32                  // key rows staged in LDS per iteration
#define NWAVE 3
#define BLOCK (NWAVE * 32)
#define KCOLS 40                  // padded stride (halves) of transposed K tile (80B, 16B-aligned)

typedef __attribute__((ext_vector_type(16))) __bf16 v16bf;
typedef __attribute__((ext_vector_type(8)))  float  v8f;

struct FragBits { uint4 lo, hi; };

// Native bf16 convert (RNE) -- lowers to the hw convert instead of integer emulation.
__device__ __forceinline__ unsigned short f2bf(float f) {
  __bf16 h = (__bf16)f;
  return __builtin_bit_cast(unsigned short, h);
}
__device__ __forceinline__ unsigned pack2(float a, float b) {
  return (unsigned)f2bf(a) | ((unsigned)f2bf(b) << 16);
}

// A-operand 16x32 bf16 fragment from row-major LDS tile.
// lane L -> row M=L%16; lanes 0-15 hold K=kbase+{0..15}, lanes 16-31 K=kbase+{16..31}.
__device__ __forceinline__ v16bf load_a_frag_lds(const unsigned short* lds, int stride,
                                                 int row, int kbase, int lane) {
  int k0 = (lane & 16) ? 16 : 0;
  const unsigned short* p = lds + row * stride + kbase + k0;
  FragBits fb;
  fb.lo = *(const uint4*)(p);        // K = k0..k0+7
  fb.hi = *(const uint4*)(p + 8);    // K = k0+8..k0+15
  return __builtin_bit_cast(v16bf, fb);
}

// B-operand 32x16 bf16 fragment: lane L -> column N=L%16; per half, 16 contiguous K values.
// "row" = the LDS row holding that B-column's data (row-major K tile / transposed tile / etc).
__device__ __forceinline__ v16bf load_b_frag_lds(const unsigned short* lds, int stride,
                                                 int row, int kbase, int lane) {
  int k0 = (lane & 16) ? 16 : 0;
  const unsigned short* p = lds + row * stride + kbase + k0;
  FragBits fb;
  fb.lo = *(const uint4*)(p);        // K = k0..k0+7
  fb.hi = *(const uint4*)(p + 8);    // K = k0+8..k0+15
  return __builtin_bit_cast(v16bf, fb);
}

// B-operand fragment built straight from global fp32 row-major W (columns of B = rows of W).
__device__ __forceinline__ v16bf load_b_frag_gW(const float* wrow, int kbase, int lane) {
  int k0 = (lane & 16) ? 16 : 0;
  const float4* p = (const float4*)(wrow + kbase + k0);
  float4 f0 = p[0], f1 = p[1], f2 = p[2], f3 = p[3];
  FragBits fb;
  fb.lo.x = pack2(f0.x, f0.y); fb.lo.y = pack2(f0.z, f0.w);
  fb.lo.z = pack2(f1.x, f1.y); fb.lo.w = pack2(f1.z, f1.w);
  fb.hi.x = pack2(f2.x, f2.y); fb.hi.y = pack2(f2.z, f2.w);
  fb.hi.z = pack2(f3.x, f3.y); fb.hi.w = pack2(f3.z, f3.w);
  return __builtin_bit_cast(v16bf, fb);
}

#define WMMA_BF16(A, Bm, Cm) \
  __builtin_amdgcn_wmma_f32_16x16x32_bf16(false, (A), false, (Bm), (short)0, (Cm), false, false)

// LDS partition (halves): qtile 48x256 | krow 32x256 | kcol 256x40 | pbuf 3x16x32
#define OFF_Q  0
#define OFF_KR (OFF_Q + QTILE * CH)        // 12288
#define OFF_KC (OFF_KR + KTILE * CH)       // 20480
#define OFF_P  (OFF_KC + CH * KCOLS)       // 30720
#define SMEM_H (OFF_P + NWAVE * 16 * 32)   // 32256 halves = 64512 B

__global__ void __launch_bounds__(BLOCK)
nonlocal_attn_kernel(const float* __restrict__ x, const float* __restrict__ Wm,
                     float* __restrict__ out) {
  __shared__ unsigned short smem[SMEM_H];
  unsigned short* qtile = smem + OFF_Q;
  unsigned short* krow  = smem + OFF_KR;
  unsigned short* kcol  = smem + OFF_KC;
  unsigned short* pbuf  = smem + OFF_P;

  const int tid  = threadIdx.x;
  const int lane = tid & 31;
  const int wave = tid >> 5;
  const int l16  = lane & 15;
  const int up8  = (lane & 16) ? 8 : 0;
  const int b    = blockIdx.x / (NSP / QTILE);
  const int n0   = (blockIdx.x % (NSP / QTILE)) * QTILE;

  const float* g = x + (size_t)b * CNE;    // [N, C] row-major view of this batch

  // ---- stage Q tile (48 x 256) as bf16: float4 loads, 8-byte LDS stores ----
  for (int i = tid; i < QTILE * CH / 4; i += BLOCK) {
    int r = i >> 6, c4 = (i & 63) * 4;
    float4 f = *(const float4*)(g + (size_t)(n0 + r) * CH + c4);
    uint2 pk; pk.x = pack2(f.x, f.y); pk.y = pack2(f.z, f.w);
    *(uint2*)(qtile + r * CH + c4) = pk;
  }

  const v8f zero8 = {0.f, 0.f, 0.f, 0.f, 0.f, 0.f, 0.f, 0.f};
  float mrow[8], lrow[8];
  v8f oacc[16];                             // 16 q-rows x 256 channels, f32
#pragma unroll
  for (int r = 0; r < 8; ++r) { mrow[r] = -1e30f; lrow[r] = 0.f; }
#pragma unroll
  for (int cc = 0; cc < 16; ++cc) oacc[cc] = zero8;

  unsigned short* mypb = pbuf + wave * (16 * 32);
  const int qb = wave * 16;                 // this wave's first local query row

#pragma unroll 1
  for (int kt = 0; kt < NSP; kt += KTILE) {
    __syncthreads();
    // stage 32-key tile, row-major AND column-major (bf16)
    for (int i = tid; i < KTILE * CH / 4; i += BLOCK) {
      int kr = i >> 6, c4 = (i & 63) * 4;
      float4 f = *(const float4*)(g + (size_t)(kt + kr) * CH + c4);
      unsigned short h0 = f2bf(f.x), h1 = f2bf(f.y), h2 = f2bf(f.z), h3 = f2bf(f.w);
      uint2 pk; pk.x = (unsigned)h0 | ((unsigned)h1 << 16);
      pk.y = (unsigned)h2 | ((unsigned)h3 << 16);
      *(uint2*)(krow + kr * CH + c4) = pk;
      kcol[(c4 + 0) * KCOLS + kr] = h0;
      kcol[(c4 + 1) * KCOLS + kr] = h1;
      kcol[(c4 + 2) * KCOLS + kr] = h2;
      kcol[(c4 + 3) * KCOLS + kr] = h3;
    }
    __syncthreads();

    // ---- S = Q K^T : 16 q-rows x 32 keys ----
    v8f s0 = zero8, s1 = zero8;
#pragma unroll
    for (int cb = 0; cb < CH; cb += 32) {
      v16bf aq = load_a_frag_lds(qtile, CH, qb + l16, cb, lane);
      v16bf b0 = load_b_frag_lds(krow, CH, l16, cb, lane);        // keys kt+0..15
      v16bf b1 = load_b_frag_lds(krow, CH, 16 + l16, cb, lane);   // keys kt+16..31
      s0 = WMMA_BF16(aq, b0, s0);
      s1 = WMMA_BF16(aq, b1, s1);
    }

    // ---- online softmax (rows live in 16-lane halves of the D layout) ----
    float nm[8], rs[8];
#pragma unroll
    for (int r = 0; r < 8; ++r) {
      float v = fmaxf(s0[r], s1[r]);
#pragma unroll
      for (int off = 8; off > 0; off >>= 1) v = fmaxf(v, __shfl_xor(v, off, 32));
      nm[r] = fmaxf(mrow[r], v);
    }
#pragma unroll
    for (int r = 0; r < 8; ++r) {
      float p0 = __expf(s0[r] - nm[r]);
      float p1 = __expf(s1[r] - nm[r]);
      s0[r] = p0; s1[r] = p1;
      float v = p0 + p1;
#pragma unroll
      for (int off = 8; off > 0; off >>= 1) v += __shfl_xor(v, off, 32);
      rs[r] = v;
    }
#pragma unroll
    for (int r = 0; r < 8; ++r) {
      float alpha = __expf(mrow[r] - nm[r]);
      lrow[r] = lrow[r] * alpha + rs[r];
      mrow[r] = nm[r];
#pragma unroll
      for (int cc = 0; cc < 16; ++cc) oacc[cc][r] *= alpha;
    }

    // ---- transpose P (D-layout -> A-layout) through per-wave LDS scratch ----
#pragma unroll
    for (int r = 0; r < 8; ++r) {
      int row = r + up8;
      mypb[row * 32 + l16]      = f2bf(s0[r]);
      mypb[row * 32 + 16 + l16] = f2bf(s1[r]);
    }
    v16bf ap = load_a_frag_lds(mypb, 32, l16, 0, lane);   // DS in-order per wave

    // ---- O += P V (V = K tile, B-columns = channels -> transposed tile) ----
#pragma unroll
    for (int cc = 0; cc < 16; ++cc) {
      v16bf bv = load_b_frag_lds(kcol, KCOLS, cc * 16 + l16, 0, lane);
      oacc[cc] = WMMA_BF16(ap, bv, oacc[cc]);
    }
  }

  __syncthreads();   // everyone done with qtile/krow/kcol

  // ---- finalize O /= l, store mapped tile bf16 (reuse qtile region) ----
  unsigned short* mtile = smem + OFF_Q;     // 48 x 256
#pragma unroll
  for (int r = 0; r < 8; ++r) {
    float inv = 1.0f / lrow[r];
    int row = qb + r + up8;
#pragma unroll
    for (int cc = 0; cc < 16; ++cc)
      mtile[row * CH + cc * 16 + l16] = f2bf(oacc[cc][r] * inv);
  }
  // each wave reads back only its own 16 rows -> per-wave DS ordering suffices

  const size_t TOT = (size_t)BATCH * CNE;
  float* finalp = out;
  float* maskp  = out + 2 * TOT;
  const float* xb = x + (size_t)b * CNE;

  // ---- mask = mapped @ W^T ; final = mask + x ----
#pragma unroll 1
  for (int ob = 0; ob < CH; ob += 16) {
    v8f acc = zero8;
    const float* wrow = Wm + (size_t)(ob + l16) * CH;
#pragma unroll
    for (int cb = 0; cb < CH; cb += 32) {
      v16bf am = load_a_frag_lds(mtile, CH, qb + l16, cb, lane);
      v16bf bw = load_b_frag_gW(wrow, cb, lane);
      acc = WMMA_BF16(am, bw, acc);
    }
    int o = ob + l16;
#pragma unroll
    for (int r = 0; r < 8; ++r) {
      int n = n0 + qb + r + up8;
      size_t idx = (size_t)b * CNE + (size_t)o * NSP + n;
      float mv = acc[r];
      maskp[idx]  = mv;
      finalp[idx] = mv + xb[(size_t)o * NSP + n];
    }
  }
}

// Second output of the tuple is x itself: plain vectorized copy.
__global__ void copy_x_kernel(const float* __restrict__ x, float* __restrict__ outx,
                              unsigned n4) {
  unsigned i = blockIdx.x * blockDim.x + threadIdx.x;
  if (i < n4) ((float4*)outx)[i] = ((const float4*)x)[i];
}

extern "C" void kernel_launch(void* const* d_in, const int* in_sizes, int n_in,
                              void* d_out, int out_size, void* d_ws, size_t ws_size,
                              hipStream_t stream) {
  (void)in_sizes; (void)n_in; (void)out_size; (void)d_ws; (void)ws_size;
  const float* x  = (const float*)d_in[0];
  const float* Wm = (const float*)d_in[1];
  float* out = (float*)d_out;

  dim3 grid(BATCH * (NSP / QTILE));                 // 384 workgroups
  nonlocal_attn_kernel<<<grid, BLOCK, 0, stream>>>(x, Wm, out);

  const size_t TOT = (size_t)BATCH * CNE;           // 4,718,592 floats
  unsigned n4 = (unsigned)(TOT / 4);                // exact multiple
  copy_x_kernel<<<(n4 + 255) / 256, 256, 0, stream>>>(x, out + TOT, n4);
}
